// DSMIL_33311766348044
// MI455X (gfx1250) — compile-verified, matrix operand order checked by reading, not run on hardware
//
#include <hip/hip_runtime.h>
#include <hip/hip_bf16.h>
#include <cstdint>

// ---------------------------------------------------------------------------
// Problem constants
// ---------------------------------------------------------------------------
#define BATCH   4
#define NBAG    16384
#define IN_DIM  1024
#define EDIM    512
#define MTOT    (BATCH * NBAG)          // 65536 rows of h

typedef __attribute__((ext_vector_type(16))) __bf16 bf16x16;
typedef __attribute__((ext_vector_type(8)))  __bf16 bf16x8;
typedef __attribute__((ext_vector_type(4)))  __bf16 bf16x4;
typedef __attribute__((ext_vector_type(8)))  float  v8f;
typedef __attribute__((ext_vector_type(4)))  int    v4i;

// ---------------------------------------------------------------------------
// CDNA5 async global->LDS path (ASYNCcnt-tracked, bypasses VGPRs).
// ---------------------------------------------------------------------------
#if defined(__HIP_DEVICE_COMPILE__)
# if __has_builtin(__builtin_amdgcn_global_load_async_to_lds_b128) && \
     __has_builtin(__builtin_amdgcn_s_wait_asynccnt)
#  define USE_ASYNC 1
# else
#  define USE_ASYNC 0
#  warning "gfx1250 async-to-LDS builtins unavailable: using sync LDS copies"
# endif
#else
# define USE_ASYNC 0
#endif

__device__ __forceinline__ void cp16_to_lds(const __bf16* __restrict__ g, __bf16* l) {
#if USE_ASYNC
    __builtin_amdgcn_global_load_async_to_lds_b128(
        (__attribute__((address_space(1))) v4i*)(uintptr_t)g,
        (__attribute__((address_space(3))) v4i*)(uint32_t)(uintptr_t)l,
        0, 0);
#else
    *(bf16x8*)l = *(const bf16x8*)g;
#endif
}

__device__ __forceinline__ void wait_async_zero() {
#if USE_ASYNC
    __builtin_amdgcn_s_wait_asynccnt(0);
#endif
}

// ---------------------------------------------------------------------------
// 0) init: zero the atomic accumulators (must rerun every call: graph replay)
// ---------------------------------------------------------------------------
__global__ __launch_bounds__(256) void init_kernel(
    unsigned long long* __restrict__ best,   // [B]
    unsigned int* __restrict__ maxbits,      // [B]
    float* __restrict__ denom,               // [B]
    float* __restrict__ attn_bag)            // [B][E]
{
    int t = threadIdx.x;
    if (t < BATCH) { best[t] = 0ull; maxbits[t] = 0u; denom[t] = 0.0f; }
    for (int i = t; i < BATCH * EDIM; i += 256) attn_bag[i] = 0.0f;
}

// ---------------------------------------------------------------------------
// 1) x fp32 -> bf16  (134 MB result: fits MI455X 192 MB global L2)
// ---------------------------------------------------------------------------
__global__ __launch_bounds__(256) void convert_x_kernel(
    const float* __restrict__ x, __bf16* __restrict__ xb)
{
    long long i = (long long)blockIdx.x * 256 + threadIdx.x;   // float4 index
    float4 v = ((const float4*)x)[i];
    bf16x4 o;
    o[0] = (__bf16)v.x; o[1] = (__bf16)v.y; o[2] = (__bf16)v.z; o[3] = (__bf16)v.w;
    ((bf16x4*)xb)[i] = o;
}

// ---------------------------------------------------------------------------
// 2) weT[e][k] = (bf16) we[k][e]   (1 MB, trivial)
// ---------------------------------------------------------------------------
__global__ __launch_bounds__(256) void wt_kernel(
    const float* __restrict__ we, __bf16* __restrict__ wtb)
{
    int t = blockIdx.x * 256 + threadIdx.x;       // 1024*512 threads
    int k = t >> 9;
    int e = t & 511;
    wtb[(size_t)e * IN_DIM + k] = (__bf16)we[(size_t)k * EDIM + e];
}

// ---------------------------------------------------------------------------
// 3) WMMA GEMM: h[m][n] = sum_k xb[m][k] * wtb[n][k] + be[n]
//    Block tile 256x128, K-slab 32, double-buffered LDS fed by async-to-LDS.
//    8 waves (wave32) in 4x2; wave tile 64x64 = 4x4 grid of 16x16 wmma tiles:
//    8 ds_load_b128 fragments -> 16 WMMAs per slab per wave.
// ---------------------------------------------------------------------------
#define BM 256
#define BN 128
#define BK 32
#define LDT 40          // LDS row stride in bf16 (80 B: 16B-aligned, bank-skewed)
#define NKSLAB (IN_DIM / BK)

__global__ __launch_bounds__(256) void gemm_h_kernel(
    const __bf16* __restrict__ xb,    // [MTOT][IN_DIM]
    const __bf16* __restrict__ wtb,   // [EDIM][IN_DIM]
    const float*  __restrict__ be,    // [EDIM]
    float* __restrict__ h)            // [MTOT][EDIM]
{
    __shared__ __align__(16) __bf16 As[2][BM * LDT];   // 40960 B
    __shared__ __align__(16) __bf16 Bs[2][BN * LDT];   // 20480 B

    const int tid  = threadIdx.x;
    const int lane = tid & 31;
    const int wid  = tid >> 5;        // 0..7
    const int wm   = wid >> 1;        // 0..3  -> rows wm*64 .. +63
    const int wn   = wid & 1;         // 0..1  -> cols wn*64 .. +63

    const int m0 = blockIdx.x * BM;
    const int n0 = blockIdx.y * BN;

    // cooperative-load coords: rows x 32 bf16; 16B units
    const int lr0 = tid >> 2;               // base row (0..63)
    const int ls  = (tid & 3) << 3;         // bf16 offset within row: 0,8,16,24

    v8f acc[4][4];
#pragma unroll
    for (int i = 0; i < 4; ++i)
#pragma unroll
        for (int j = 0; j < 4; ++j) acc[i][j] = (v8f)(0.0f);

    const int frow = lane & 15;             // fragment row (M for A, N for B)
    const int k0l  = (lane >> 4) << 3;      // lane-half K offset: 0 or 8

    // prologue: stage slab 0 into buffer 0
#pragma unroll
    for (int i = 0; i < 4; ++i) {           // A: 256 rows -> 1024 units
        int r = lr0 + i * 64;
        cp16_to_lds(&xb[(size_t)(m0 + r) * IN_DIM + ls], &As[0][r * LDT + ls]);
    }
#pragma unroll
    for (int i = 0; i < 2; ++i) {           // B: 128 rows -> 512 units
        int r = lr0 + i * 64;
        cp16_to_lds(&wtb[(size_t)(n0 + r) * IN_DIM + ls], &Bs[0][r * LDT + ls]);
    }

    for (int kb = 0; kb < NKSLAB; ++kb) {
        const int cur = kb & 1;
        wait_async_zero();          // my async writes into As/Bs[cur] landed
        __syncthreads();            // everyone's landed; prev buffer fully read

        if (kb + 1 < NKSLAB) {      // stream next slab into the other buffer
            const int nkb = (kb + 1) * BK;
#pragma unroll
            for (int i = 0; i < 4; ++i) {
                int r = lr0 + i * 64;
                cp16_to_lds(&xb[(size_t)(m0 + r) * IN_DIM + nkb + ls],
                            &As[cur ^ 1][r * LDT + ls]);
            }
#pragma unroll
            for (int i = 0; i < 2; ++i) {
                int r = lr0 + i * 64;
                cp16_to_lds(&wtb[(size_t)(n0 + r) * IN_DIM + nkb + ls],
                            &Bs[cur ^ 1][r * LDT + ls]);
            }
        }

        // fragments from the current buffer: 4 A + 4 B, then 16 WMMAs
        bf16x16 af[4], bfm[4];
#pragma unroll
        for (int i = 0; i < 4; ++i) {
            const __bf16* src = &As[cur][(wm * 64 + i * 16 + frow) * LDT + k0l];
            bf16x8 lo = *(const bf16x8*)src;
            bf16x8 hi = *(const bf16x8*)(src + 16);
#pragma unroll
            for (int e = 0; e < 8; ++e) { af[i][e] = lo[e]; af[i][8 + e] = hi[e]; }
        }
#pragma unroll
        for (int j = 0; j < 4; ++j) {
            const __bf16* src = &Bs[cur][(wn * 64 + j * 16 + frow) * LDT + k0l];
            bf16x8 lo = *(const bf16x8*)src;
            bf16x8 hi = *(const bf16x8*)(src + 16);
#pragma unroll
            for (int e = 0; e < 8; ++e) { bfm[j][e] = lo[e]; bfm[j][8 + e] = hi[e]; }
        }
#pragma unroll
        for (int i = 0; i < 4; ++i)
#pragma unroll
            for (int j = 0; j < 4; ++j)
                acc[i][j] = __builtin_amdgcn_wmma_f32_16x16x32_bf16(
                    false, af[i], false, bfm[j], (short)0, acc[i][j], false, false);
    }

    // C/D layout: lane -> col = lane&15 ; VGPR v -> row = v + 8*(lane>=16)
    const int coln = lane & 15;
    const int rowh = (lane >> 4) << 3;
#pragma unroll
    for (int j = 0; j < 4; ++j) {
        int n = n0 + wn * 64 + j * 16 + coln;
        float bias = be[n];
#pragma unroll
        for (int i = 0; i < 4; ++i) {
            int mbase = m0 + wm * 64 + i * 16 + rowh;
#pragma unroll
            for (int v = 0; v < 8; ++v)
                h[(size_t)(mbase + v) * EDIM + n] = acc[i][j][v] + bias;
        }
    }
}

// ---------------------------------------------------------------------------
// 4) instance logits (C=2), per-bag argmax of max-logit via packed u64 atomicMax
// ---------------------------------------------------------------------------
__device__ __forceinline__ unsigned int ordered_bits(float f) {
    unsigned int u = __float_as_uint(f);
    return (u & 0x80000000u) ? ~u : (u | 0x80000000u);
}

__global__ __launch_bounds__(256) void inst_argmax_kernel(
    const float* __restrict__ h, const float* __restrict__ wi,
    const float* __restrict__ bi, unsigned long long* __restrict__ best)
{
    __shared__ float w0[EDIM], w1[EDIM];
    __shared__ unsigned long long wbest[8];
    int t = threadIdx.x, lane = t & 31, wid = t >> 5;
    for (int i = t; i < EDIM; i += 256) { w0[i] = wi[i * 2 + 0]; w1[i] = wi[i * 2 + 1]; }
    __syncthreads();
    int b = blockIdx.y;
    int rowbase = blockIdx.x * 128;
    float bi0 = bi[0], bi1 = bi[1];
    unsigned long long mybest = 0ull;
    for (int r = wid; r < 128; r += 8) {
        int n = rowbase + r;
        const float* hr = h + ((size_t)b * NBAG + n) * EDIM;
        float a0 = 0.f, a1 = 0.f;
        for (int e = lane; e < EDIM; e += 32) {
            float hv = hr[e];
            a0 += hv * w0[e]; a1 += hv * w1[e];
        }
        for (int off = 16; off > 0; off >>= 1) {
            a0 += __shfl_xor(a0, off, 32);
            a1 += __shfl_xor(a1, off, 32);
        }
        if (lane == 0) {
            float s = fmaxf(a0 + bi0, a1 + bi1);
            unsigned long long packed =
                ((unsigned long long)ordered_bits(s) << 32) | (unsigned int)n;
            if (packed > mybest) mybest = packed;
        }
    }
    if (lane == 0) wbest[wid] = mybest;
    __syncthreads();
    if (t == 0) {
        unsigned long long m = wbest[0];
        for (int i = 1; i < 8; ++i) m = m > wbest[i] ? m : wbest[i];
        atomicMax(&best[b], m);
    }
}

// ---------------------------------------------------------------------------
// 5) critical, q = crit@wq+bq, p = (wk@q)/sqrt(E), c = (q.bk)/sqrt(E)
// ---------------------------------------------------------------------------
__global__ __launch_bounds__(256) void qp_kernel(
    const float* __restrict__ h, const float* __restrict__ wq,
    const float* __restrict__ bq, const float* __restrict__ wk,
    const float* __restrict__ bk, const unsigned long long* __restrict__ best,
    float* __restrict__ crit, float* __restrict__ pvec, float* __restrict__ cscal)
{
    __shared__ float cs[EDIM];
    __shared__ float qs[EDIM];
    __shared__ float red[256];
    int b = blockIdx.x, t = threadIdx.x;
    int idx = (int)(best[b] & 0xFFFFFFFFull);
    const float* hr = h + ((size_t)b * NBAG + idx) * EDIM;
    for (int e = t; e < EDIM; e += 256) { float v = hr[e]; cs[e] = v; crit[b * EDIM + e] = v; }
    __syncthreads();
    for (int f = t; f < EDIM; f += 256) {
        float a = bq[f];
        for (int e = 0; e < EDIM; ++e) a += cs[e] * wq[(size_t)e * EDIM + f];
        qs[f] = a;
    }
    __syncthreads();
    const float scale = rsqrtf((float)EDIM);
    for (int e = t; e < EDIM; e += 256) {
        float a = 0.f;
        for (int f = 0; f < EDIM; ++f) a += wk[(size_t)e * EDIM + f] * qs[f];
        pvec[b * EDIM + e] = a * scale;
    }
    float a = 0.f;
    for (int f = t; f < EDIM; f += 256) a += qs[f] * bk[f];
    red[t] = a;
    __syncthreads();
    for (int s = 128; s > 0; s >>= 1) { if (t < s) red[t] += red[t + s]; __syncthreads(); }
    if (t == 0) cscal[b] = red[0] * scale;
}

// ---------------------------------------------------------------------------
// 6) scores[b][n] = h[b,n].p[b] + c[b]  (already 1/sqrt(E)-scaled); track max
// ---------------------------------------------------------------------------
__global__ __launch_bounds__(256) void score_kernel(
    const float* __restrict__ h, const float* __restrict__ pvec,
    const float* __restrict__ cscal, float* __restrict__ scores,
    unsigned int* __restrict__ maxbits)
{
    __shared__ float ps[EDIM];
    __shared__ float wmax[8];
    int b = blockIdx.y, t = threadIdx.x, lane = t & 31, wid = t >> 5;
    for (int e = t; e < EDIM; e += 256) ps[e] = pvec[b * EDIM + e];
    __syncthreads();
    float c = cscal[b];
    float mymax = -3.4e38f;
    int rowbase = blockIdx.x * 256;
    for (int r = wid; r < 256; r += 8) {
        int n = rowbase + r;
        const float* hr = h + ((size_t)b * NBAG + n) * EDIM;
        float a = 0.f;
        for (int e = lane; e < EDIM; e += 32) a += hr[e] * ps[e];
        for (int off = 16; off > 0; off >>= 1) a += __shfl_xor(a, off, 32);
        if (lane == 0) {
            a += c;
            scores[b * NBAG + n] = a;
            mymax = fmaxf(mymax, a);
        }
    }
    if (lane == 0) wmax[wid] = mymax;
    __syncthreads();
    if (t == 0) {
        float m = wmax[0];
        for (int i = 1; i < 8; ++i) m = fmaxf(m, wmax[i]);
        atomicMax(&maxbits[b], ordered_bits(m));
    }
}

// ---------------------------------------------------------------------------
// 7) w = exp(s - max); denom += sum(w); attn_bag[e] += sum_n w*h[n][e]
// ---------------------------------------------------------------------------
__global__ __launch_bounds__(256) void attn_kernel(
    const float* __restrict__ h, const float* __restrict__ scores,
    const unsigned int* __restrict__ maxbits, float* __restrict__ denom,
    float* __restrict__ attn_bag)
{
    __shared__ float wl[128];
    __shared__ float red[256];
    int b = blockIdx.y, t = threadIdx.x;
    unsigned int u = maxbits[b];
    float mx = __uint_as_float((u & 0x80000000u) ? (u & 0x7FFFFFFFu) : ~u);
    int rowbase = blockIdx.x * 128;
    float lsum = 0.f;
    if (t < 128) {
        float w = __expf(scores[b * NBAG + rowbase + t] - mx);
        wl[t] = w;
        lsum = w;
    }
    red[t] = lsum;
    __syncthreads();
    for (int s = 128; s > 0; s >>= 1) { if (t < s) red[t] += red[t + s]; __syncthreads(); }
    if (t == 0) atomicAdd(&denom[b], red[0]);
    float a0 = 0.f, a1 = 0.f;
    for (int r = 0; r < 128; ++r) {
        const float* hr = h + ((size_t)b * NBAG + rowbase + r) * EDIM;
        float w = wl[r];
        a0 += w * hr[t];
        a1 += w * hr[t + 256];
    }
    atomicAdd(&attn_bag[b * EDIM + t], a0);
    atomicAdd(&attn_bag[b * EDIM + t + 256], a1);
}

// ---------------------------------------------------------------------------
// 8) out[b] = [critical, attn_bag/denom] @ wb + bb
// ---------------------------------------------------------------------------
__global__ __launch_bounds__(256) void final_kernel(
    const float* __restrict__ crit, const float* __restrict__ attn_bag,
    const float* __restrict__ denom, const float* __restrict__ wb,
    const float* __restrict__ bb, float* __restrict__ out)
{
    __shared__ float r0[256], r1[256];
    int b = blockIdx.x, t = threadIdx.x;
    float inv = 1.0f / denom[b];
    float a0 = 0.f, a1 = 0.f;
    for (int j = t; j < 2 * EDIM; j += 256) {
        float v = (j < EDIM) ? crit[b * EDIM + j] : attn_bag[b * EDIM + (j - EDIM)] * inv;
        a0 += v * wb[j * 2 + 0];
        a1 += v * wb[j * 2 + 1];
    }
    r0[t] = a0; r1[t] = a1;
    __syncthreads();
    for (int s = 128; s > 0; s >>= 1) {
        if (t < s) { r0[t] += r0[t + s]; r1[t] += r1[t + s]; }
        __syncthreads();
    }
    if (t == 0) { out[b * 2 + 0] = r0[0] + bb[0]; out[b * 2 + 1] = r1[0] + bb[1]; }
}

// ---------------------------------------------------------------------------
// host launcher
// ---------------------------------------------------------------------------
extern "C" void kernel_launch(void* const* d_in, const int* in_sizes, int n_in,
                              void* d_out, int out_size, void* d_ws, size_t ws_size,
                              hipStream_t stream) {
    const float* x  = (const float*)d_in[0];
    const float* we = (const float*)d_in[1];
    const float* be = (const float*)d_in[2];
    const float* wi = (const float*)d_in[3];
    const float* bi = (const float*)d_in[4];
    const float* wq = (const float*)d_in[5];
    const float* bq = (const float*)d_in[6];
    const float* wk = (const float*)d_in[7];
    const float* bk = (const float*)d_in[8];
    const float* wb = (const float*)d_in[9];
    const float* bb = (const float*)d_in[10];
    float* out = (float*)d_out;

    // workspace carve-up (256B aligned)
    char* ws = (char*)d_ws;
    size_t off = 0;
    auto carve = [&](size_t bytes) {
        char* p = ws + off;
        off = (off + bytes + 255) & ~(size_t)255;
        return p;
    };
    __bf16* xb        = (__bf16*)carve((size_t)MTOT * IN_DIM * 2);   // 134 MB
    __bf16* wtb       = (__bf16*)carve((size_t)EDIM * IN_DIM * 2);   // 1 MB
    float*  h         = (float*) carve((size_t)MTOT * EDIM * 4);     // 134 MB
    float*  scores    = (float*) carve((size_t)MTOT * 4);            // 256 KB
    unsigned long long* best = (unsigned long long*)carve(BATCH * 8);
    unsigned int* maxbits    = (unsigned int*)carve(BATCH * 4);
    float* denom     = (float*)carve(BATCH * 4);
    float* attn_bag  = (float*)carve(BATCH * EDIM * 4);
    float* pvec      = (float*)carve(BATCH * EDIM * 4);
    float* cscal     = (float*)carve(BATCH * 4);
    float* crit      = (float*)carve(BATCH * EDIM * 4);

    init_kernel<<<1, 256, 0, stream>>>(best, maxbits, denom, attn_bag);
    convert_x_kernel<<<(MTOT * (IN_DIM / 4)) / 256, 256, 0, stream>>>(x, xb);
    wt_kernel<<<(IN_DIM * EDIM) / 256, 256, 0, stream>>>(we, wtb);

    gemm_h_kernel<<<dim3(MTOT / BM, EDIM / BN), 256, 0, stream>>>(xb, wtb, be, h);

    inst_argmax_kernel<<<dim3(NBAG / 128, BATCH), 256, 0, stream>>>(h, wi, bi, best);
    qp_kernel<<<BATCH, 256, 0, stream>>>(h, wq, bq, wk, bk, best, crit, pvec, cscal);
    score_kernel<<<dim3(NBAG / 256, BATCH), 256, 0, stream>>>(h, pvec, cscal, scores, maxbits);
    attn_kernel<<<dim3(NBAG / 128, BATCH), 256, 0, stream>>>(h, scores, maxbits, denom, attn_bag);
    final_kernel<<<BATCH, 256, 0, stream>>>(crit, attn_bag, denom, wb, bb, out);

    (void)in_sizes; (void)n_in; (void)out_size; (void)ws_size;
}
